// LSTM_quantized_cell_80058190397499
// MI455X (gfx1250) — compile-verified, matrix operand order checked by reading, not run on hardware
//
#include <hip/hip_runtime.h>
#include <hip/hip_bf16.h>

typedef __attribute__((ext_vector_type(16))) __bf16 v16bf;
typedef __attribute__((ext_vector_type(8)))  float  v8f;
typedef __attribute__((ext_vector_type(4)))  unsigned int v4u;
typedef __attribute__((ext_vector_type(4)))  int   v4i;
typedef __attribute__((ext_vector_type(8)))  int   v8i;

#define BDIM 4096
#define HDIM 1024
#define KDIM 1024
#define NDIM 4096   // 4*H

#define BM 128
#define BN 128
#define BK 32
#define NK (KDIM / BK)

#if __has_builtin(__builtin_amdgcn_tensor_load_to_lds)
#define HAVE_TDM 1
#endif
#if __has_include(<hip/amd_detail/amd_gfx1250_TDM.h>)
#define TDM_6ARG 1
#endif

// ---------------- fp32 -> bf16 conversion ----------------
__global__ __launch_bounds__(256) void cvt_bf16(const float* __restrict__ src,
                                                __bf16* __restrict__ dst, int n) {
    int i = (blockIdx.x * 256 + threadIdx.x) * 4;
    if (i + 3 < n) {
        float4 v = *(const float4*)(src + i);
        dst[i + 0] = (__bf16)v.x;
        dst[i + 1] = (__bf16)v.y;
        dst[i + 2] = (__bf16)v.z;
        dst[i + 3] = (__bf16)v.w;
    }
}

#ifdef HAVE_TDM
// Issue one TDM 2D tile load (bf16 elements) into LDS.
// D# per CDNA5 ISA ch.8: group0 = count/lds/global/type, group1 = dims+strides.
__device__ __forceinline__ void tdm_load_tile_bf16(
    unsigned lds_addr, unsigned long long gaddr,
    unsigned tensor_d0, unsigned tensor_d1,
    unsigned long long stride0_elems,
    unsigned tile_d0, unsigned tile_d1)
{
    v4u g0;
    g0[0] = 1u;                                              // count=1, user D#
    g0[1] = lds_addr;                                        // lds_addr [63:32]
    g0[2] = (unsigned)(gaddr & 0xffffffffull);               // global_addr[31:0]
    g0[3] = (unsigned)((gaddr >> 32) & 0x01ffffffull)        // global_addr[56:32]
          | (2u << 30);                                      // type = 2 ("image")
    v8i g1;
    g1[0] = (int)(1u << 16);                                 // data_size=1 -> 2 bytes
    g1[1] = (int)((tensor_d0 & 0xffffu) << 16);              // tensor_dim0[15:0] @ [63:48]
    g1[2] = (int)((tensor_d0 >> 16) | ((tensor_d1 & 0xffffu) << 16)); // d0[31:16], d1[15:0]
    g1[3] = (int)((tensor_d1 >> 16) | (tile_d0 << 16));      // d1[31:16], tile_dim0 @ [127:112]
    g1[4] = (int)(tile_d1 & 0xffffu);                        // tile_dim1 @ [143:128]; tile_dim2=0
    g1[5] = (int)(unsigned)(stride0_elems & 0xffffffffull);  // dim0_stride[31:0]
    g1[6] = (int)(unsigned)((stride0_elems >> 32) & 0xffffull); // dim0_stride[47:32]
    g1[7] = 0;
    v4i gz = {0, 0, 0, 0};
#ifdef TDM_6ARG
    v8i gz8 = {0, 0, 0, 0, 0, 0, 0, 0};
    __builtin_amdgcn_tensor_load_to_lds(g0, g1, gz, gz, gz8, 0);
#else
    __builtin_amdgcn_tensor_load_to_lds(g0, g1, gz, gz, 0);
#endif
}
#endif

// ---------------- bf16 WMMA GEMM:  P = A(BxK) * W(NxK)^T ----------------
// blockIdx.z = 0: Pi = x * w_ih^T ;  1: Ph = h * w_hh^T
__global__ __launch_bounds__(256) void gemm_bf16(
    const __bf16* __restrict__ Ax, const __bf16* __restrict__ Ah,
    const __bf16* __restrict__ Wi, const __bf16* __restrict__ Wh,
    float* __restrict__ Pi, float* __restrict__ Ph)
{
    const __bf16* __restrict__ A = blockIdx.z ? Ah : Ax;
    const __bf16* __restrict__ W = blockIdx.z ? Wh : Wi;
    float* __restrict__ P        = blockIdx.z ? Ph : Pi;

    const int t    = threadIdx.x;
    const int m0   = blockIdx.x * BM;
    const int n0   = blockIdx.y * BN;
    const int wave = t >> 5;
    const int lane = t & 31;
    const int wm   = (wave & 1) * 64;   // 2 waves along M, 64 rows each
    const int wn   = (wave >> 1) * 32;  // 4 waves along N, 32 cols each
    const int fr   = lane & 15;         // fragment row
    const int fk   = (lane >> 4) * 8;   // fragment K base (bf16 units)

    v8f acc[4][2];
    const v8f vzero = {0.f, 0.f, 0.f, 0.f, 0.f, 0.f, 0.f, 0.f};
#pragma unroll
    for (int i = 0; i < 4; ++i)
#pragma unroll
        for (int j = 0; j < 2; ++j) acc[i][j] = vzero;

    union FragB { uint4 u[2]; v16bf v; };

#ifdef HAVE_TDM
    // ---- TDM-staged, double-buffered pipeline ----
    __shared__ alignas(16) __bf16 sA[2][BM * BK];
    __shared__ alignas(16) __bf16 sB[2][BN * BK];

    const unsigned long long abase =
        (unsigned long long)(size_t)A + (unsigned long long)m0 * KDIM * 2ull;
    const unsigned long long wbase =
        (unsigned long long)(size_t)W + (unsigned long long)n0 * KDIM * 2ull;
    const unsigned ldsA0 = (unsigned)(size_t)&sA[0][0];
    const unsigned ldsA1 = (unsigned)(size_t)&sA[1][0];
    const unsigned ldsB0 = (unsigned)(size_t)&sB[0][0];
    const unsigned ldsB1 = (unsigned)(size_t)&sB[1][0];

    if (wave == 0) {   // one DMA pair per stage, issued by wave 0 only
        tdm_load_tile_bf16(ldsA0, abase, KDIM, BDIM, KDIM, BK, BM);
        tdm_load_tile_bf16(ldsB0, wbase, KDIM, NDIM, KDIM, BK, BN);
    }

    for (int ks = 0; ks < NK; ++ks) {
        const int cur = ks & 1;
        if (wave == 0) {
            if (ks + 1 < NK) {
                const unsigned long long off =
                    (unsigned long long)(ks + 1) * BK * 2ull;
                tdm_load_tile_bf16(cur ? ldsA0 : ldsA1, abase + off,
                                   KDIM, BDIM, KDIM, BK, BM);
                tdm_load_tile_bf16(cur ? ldsB0 : ldsB1, wbase + off,
                                   KDIM, NDIM, KDIM, BK, BN);
                // 4 outstanding; <=2 means stage ks's pair has completed (in-order)
                __builtin_amdgcn_s_wait_tensorcnt(2);
            } else {
                __builtin_amdgcn_s_wait_tensorcnt(0);
            }
        }
        __syncthreads();   // publish stage ks tiles to all waves

        FragB fa[4], fb[2];
#pragma unroll
        for (int i = 0; i < 4; ++i) {
            int row = wm + i * 16 + fr;
            fa[i].u[0] = *(const uint4*)&sA[cur][row * BK + fk];
            fa[i].u[1] = *(const uint4*)&sA[cur][row * BK + 16 + fk];
        }
#pragma unroll
        for (int j = 0; j < 2; ++j) {
            int row = wn + j * 16 + fr;
            fb[j].u[0] = *(const uint4*)&sB[cur][row * BK + fk];
            fb[j].u[1] = *(const uint4*)&sB[cur][row * BK + 16 + fk];
        }
#pragma unroll
        for (int i = 0; i < 4; ++i)
#pragma unroll
            for (int j = 0; j < 2; ++j)
                acc[i][j] = __builtin_amdgcn_wmma_f32_16x16x32_bf16(
                    false, fa[i].v, false, fb[j].v, (short)0, acc[i][j],
                    false, false);
        __syncthreads();   // all waves done with buf `cur` before it is re-DMA'd
    }
#else
    // ---- fallback: manual global->VGPR->LDS staging ----
    __shared__ alignas(16) __bf16 sA[1][BM * BK];
    __shared__ alignas(16) __bf16 sB[1][BN * BK];

    for (int k0 = 0; k0 < KDIM; k0 += BK) {
#pragma unroll
        for (int r = 0; r < 2; ++r) {
            int id  = t + r * 256;
            int row = id >> 2;
            int col = (id & 3) * 8;
            *(uint4*)&sA[0][row * BK + col] =
                *(const uint4*)&A[(size_t)(m0 + row) * KDIM + k0 + col];
            *(uint4*)&sB[0][row * BK + col] =
                *(const uint4*)&W[(size_t)(n0 + row) * KDIM + k0 + col];
        }
        __syncthreads();
        FragB fa[4], fb[2];
#pragma unroll
        for (int i = 0; i < 4; ++i) {
            int row = wm + i * 16 + fr;
            fa[i].u[0] = *(const uint4*)&sA[0][row * BK + fk];
            fa[i].u[1] = *(const uint4*)&sA[0][row * BK + 16 + fk];
        }
#pragma unroll
        for (int j = 0; j < 2; ++j) {
            int row = wn + j * 16 + fr;
            fb[j].u[0] = *(const uint4*)&sB[0][row * BK + fk];
            fb[j].u[1] = *(const uint4*)&sB[0][row * BK + 16 + fk];
        }
#pragma unroll
        for (int i = 0; i < 4; ++i)
#pragma unroll
            for (int j = 0; j < 2; ++j)
                acc[i][j] = __builtin_amdgcn_wmma_f32_16x16x32_bf16(
                    false, fa[i].v, false, fb[j].v, (short)0, acc[i][j],
                    false, false);
        __syncthreads();
    }
#endif

    // C/D layout: VGPR e, lanes 0-15 -> M=e, lanes 16-31 -> M=8+e; N = lane&15
    const int cn = lane & 15;
    const int cm = (lane >> 4) * 8;
#pragma unroll
    for (int i = 0; i < 4; ++i)
#pragma unroll
        for (int j = 0; j < 2; ++j)
#pragma unroll
            for (int e = 0; e < 8; ++e) {
                int row = m0 + wm + i * 16 + cm + e;
                int col = n0 + wn + j * 16 + cn;
                P[(size_t)row * NDIM + col] = acc[i][j][e];
            }
}

// ---------------- fused LN + gate epilogue, one block per batch row ----------------
__device__ __forceinline__ float blk_sum(float v, float* red) {
#pragma unroll
    for (int o = 16; o > 0; o >>= 1) v += __shfl_down(v, o, 32);
    const int wave = threadIdx.x >> 5, lane = threadIdx.x & 31;
    if (lane == 0) red[wave] = v;
    __syncthreads();
    float s;
    if (wave == 0) {
        s = (lane < 8) ? red[lane] : 0.f;
#pragma unroll
        for (int o = 4; o > 0; o >>= 1) s += __shfl_down(s, o, 32);
        if (lane == 0) red[0] = s;
    }
    __syncthreads();
    s = red[0];
    __syncthreads();
    return s;
}

__global__ __launch_bounds__(256) void lnlstm_epilogue(
    const float* __restrict__ Pi, const float* __restrict__ Ph,
    const float* __restrict__ cin, const float* __restrict__ bih,
    const float* __restrict__ gamma, const float* __restrict__ beta,
    float* __restrict__ h_new, float* __restrict__ c_new)
{
    const int H = HDIM;
    const int b = blockIdx.x;
    const int t = threadIdx.x;
    __shared__ float gate[4][HDIM];
    __shared__ float red[8];

    for (int g = 0; g < 4; ++g) {
        float vih[4], vhh[4];
        float si = 0.f, qi = 0.f, sh = 0.f, qh = 0.f;
#pragma unroll
        for (int e = 0; e < 4; ++e) {
            int idx = t + e * 256;
            vih[e] = Pi[(size_t)b * NDIM + g * H + idx];
            vhh[e] = Ph[(size_t)b * NDIM + g * H + idx];
            si += vih[e]; qi += vih[e] * vih[e];
            sh += vhh[e]; qh += vhh[e] * vhh[e];
        }
        si = blk_sum(si, red); qi = blk_sum(qi, red);
        sh = blk_sum(sh, red); qh = blk_sum(qh, red);
        const float mi = si * (1.f / H), mh = sh * (1.f / H);
        const float ri = rsqrtf(qi * (1.f / H) - mi * mi + 1e-5f);
        const float rh = rsqrtf(qh * (1.f / H) - mh * mh + 1e-5f);
#pragma unroll
        for (int e = 0; e < 4; ++e) {
            int idx = t + e * 256;
            float gih = gamma[(2 * g) * H + idx] * (vih[e] - mi) * ri +
                        beta[(2 * g) * H + idx];
            float ghh = gamma[(2 * g + 1) * H + idx] * (vhh[e] - mh) * rh +
                        beta[(2 * g + 1) * H + idx];
            gate[g][idx] = gih + ghh + bih[g * H + idx];
        }
    }
    __syncthreads();

    float cn[4], og[4];
    float sc = 0.f, qc = 0.f;
#pragma unroll
    for (int e = 0; e < 4; ++e) {
        int idx = t + e * 256;
        float ig = gate[0][idx], fg = gate[1][idx], ag = gate[2][idx];
        og[e] = gate[3][idx];
        float v = (1.f / (1.f + __expf(-fg))) * cin[(size_t)b * H + idx] +
                  (1.f / (1.f + __expf(-ig))) * tanhf(ag);
        cn[e] = v;
        c_new[(size_t)b * H + idx] = v;
        sc += v; qc += v * v;
    }
    sc = blk_sum(sc, red);
    qc = blk_sum(qc, red);
    const float mc = sc * (1.f / H);
    const float rc = rsqrtf(qc * (1.f / H) - mc * mc + 1e-5f);
#pragma unroll
    for (int e = 0; e < 4; ++e) {
        int idx = t + e * 256;
        float lnc = gamma[8 * H + idx] * (cn[e] - mc) * rc + beta[8 * H + idx];
        h_new[(size_t)b * H + idx] = (1.f / (1.f + __expf(-og[e]))) * tanhf(lnc);
    }
}

// ---------------- launch ----------------
extern "C" void kernel_launch(void* const* d_in, const int* in_sizes, int n_in,
                              void* d_out, int out_size, void* d_ws, size_t ws_size,
                              hipStream_t stream) {
    const float* x     = (const float*)d_in[0];
    const float* h     = (const float*)d_in[1];
    const float* c     = (const float*)d_in[2];
    const float* w_ih  = (const float*)d_in[3];
    const float* w_hh  = (const float*)d_in[4];
    const float* b_ih  = (const float*)d_in[5];
    const float* gamma = (const float*)d_in[6];
    const float* beta  = (const float*)d_in[7];

    char* ws = (char*)d_ws;
    size_t off = 0;
    __bf16* xbf  = (__bf16*)(ws + off); off += (size_t)BDIM * KDIM * 2;
    __bf16* hbf  = (__bf16*)(ws + off); off += (size_t)BDIM * HDIM * 2;
    __bf16* wibf = (__bf16*)(ws + off); off += (size_t)NDIM * KDIM * 2;
    __bf16* whbf = (__bf16*)(ws + off); off += (size_t)NDIM * HDIM * 2;
    float*  Pi   = (float*)(ws + off);  off += (size_t)BDIM * NDIM * 4;
    float*  Ph   = (float*)(ws + off);

    const int nA = BDIM * KDIM;   // 4M
    const int nW = NDIM * KDIM;   // 4M
    cvt_bf16<<<nA / 1024, 256, 0, stream>>>(x,    xbf,  nA);
    cvt_bf16<<<nA / 1024, 256, 0, stream>>>(h,    hbf,  nA);
    cvt_bf16<<<nW / 1024, 256, 0, stream>>>(w_ih, wibf, nW);
    cvt_bf16<<<nW / 1024, 256, 0, stream>>>(w_hh, whbf, nW);

    gemm_bf16<<<dim3(BDIM / BM, NDIM / BN, 2), 256, 0, stream>>>(
        xbf, hbf, wibf, whbf, Pi, Ph);

    float* h_new = (float*)d_out;
    float* c_new = h_new + (size_t)BDIM * HDIM;
    lnlstm_epilogue<<<BDIM, 256, 0, stream>>>(Pi, Ph, c, b_ih, gamma, beta,
                                              h_new, c_new);
}
